// CrossAttention_1752346657256
// MI455X (gfx1250) — matrix operand order repaired and checked by reading, not compiled, not run on hardware
//
#include <hip/hip_runtime.h>

typedef __attribute__((ext_vector_type(16))) __bf16 v16bf;
typedef __attribute__((ext_vector_type(8)))  float  v8f;
typedef __attribute__((ext_vector_type(4)))  int    v4i;

#define HWN 4096
#define CCH 256
#define DQK 32
#define L2E 1.44269504f

#define KC   64          // keys per attention step
#define SVLD 72          // LDS stride of V chunk rows (64 + 8 pad)
#define SKLD 40          // LDS stride of K chunk rows (32 + 8 pad)

#if defined(__gfx1250__) && defined(__has_builtin)
#if __has_builtin(__builtin_amdgcn_global_load_async_to_lds_b128) && \
    __has_builtin(__builtin_amdgcn_s_wait_asynccnt)
#define USE_ASYNC 1
#endif
#endif
#ifndef USE_ASYNC
#define USE_ASYNC 0
#endif

#if USE_ASYNC
typedef __attribute__((address_space(1))) v4i gv4i;
typedef __attribute__((address_space(3))) v4i lv4i;
#endif

union Frag {
  v16bf v;
  uint4 q[2];
  unsigned u[8];
};

// A-fragment: [16 x 32] 16-bit, row-major source with leading dim ld.
// lanes 0-15: row=lane,    K = 0..15 (contiguous)
// lanes16-31: row=lane-16, K = 16..31 (contiguous)
__device__ inline Frag load_afrag(const __bf16* base, int row, int ld, int lane) {
  Frag f;
  const __bf16* p = base + (size_t)row * ld + ((lane & 16) ? 16 : 0);
  f.q[0] = *(const uint4*)(p);
  f.q[1] = *(const uint4*)(p + 8);
  return f;
}

// B-fragment: [32 x 16] 16-bit from [N,K] row-major source (ld = K stride).
// lanes 0-15: n=lane,    K = 0..15 (contiguous)
// lanes16-31: n=lane-16, K = 16..31 (contiguous)
__device__ inline Frag load_bfrag(const __bf16* base, int row, int ld, int lane) {
  Frag f;
  const __bf16* p = base + (size_t)row * ld + ((lane & 16) ? 16 : 0);
  f.q[0] = *(const uint4*)(p);
  f.q[1] = *(const uint4*)(p + 8);
  return f;
}

__device__ inline v8f wmma_bf16(const Frag& a, const Frag& b, v8f c) {
  return __builtin_amdgcn_wmma_f32_16x16x32_bf16(false, a.v, false, b.v,
                                                 (short)0, c, false, false);
}

__device__ inline unsigned pack_bf16(float a, float b) {
  union { __bf16 h[2]; unsigned u; } t;
  t.h[0] = (__bf16)a; t.h[1] = (__bf16)b;
  return t.u;
}

// 16-byte copy global -> LDS; async (ASYNCcnt-tracked) when available.
__device__ inline void cp16_g2l(__bf16* lds_dst, const __bf16* gsrc) {
#if USE_ASYNC
  __builtin_amdgcn_global_load_async_to_lds_b128(
      (gv4i*)gsrc, (lv4i*)lds_dst, 0, 0);
#else
  *(uint4*)lds_dst = *(const uint4*)gsrc;
#endif
}

__device__ inline void wait_async_done() {
#if USE_ASYNC
  __builtin_amdgcn_s_wait_asynccnt(0);
#endif
}

// ---------------- pass 0: transpose + convert [B,C,N] f32 -> [B,N,C] bf16
__global__ __launch_bounds__(256) void transpose_cvt_kernel(
    const float* __restrict__ src, __bf16* __restrict__ dst, int Cdim, int Ndim)
{
  __shared__ float tile[32][33];
  int b = blockIdx.z;
  int n0 = blockIdx.x * 32, c0 = blockIdx.y * 32;
  int tx = threadIdx.x & 31, ty = threadIdx.x >> 5;  // ty in 0..7
  const float* s = src + (size_t)b * Cdim * Ndim;
  for (int i = 0; i < 32; i += 8)
    tile[ty + i][tx] = s[(size_t)(c0 + ty + i) * Ndim + (n0 + tx)];
  __syncthreads();
  __bf16* d = dst + (size_t)b * Ndim * Cdim;
  for (int i = 0; i < 32; i += 8)
    d[(size_t)(n0 + ty + i) * Cdim + (c0 + tx)] = (__bf16)tile[tx][ty + i];
}

__global__ void cvt_bf16_kernel(const float* __restrict__ s, __bf16* __restrict__ d, int n) {
  int i = blockIdx.x * blockDim.x + threadIdx.x;
  if (i < n) d[i] = (__bf16)s[i];
}

// ---------------- pass 1: Q/K projection  out[b,n,o] = inT[b,n,:] . W[o,:] + bias[o]
__global__ __launch_bounds__(256) void qk_proj_kernel(
    const __bf16* __restrict__ inT,  // [B, HW, 256]
    const __bf16* __restrict__ W,    // [32, 256]
    const float* __restrict__ bias,  // [32]
    __bf16* __restrict__ outp)       // [B, HW, 32]
{
  int b = blockIdx.y;
  int w = threadIdx.x >> 5, lane = threadIdx.x & 31;
  int n0 = blockIdx.x * 128 + w * 16;
  int row = lane & 15;
  const __bf16* A = inT + ((size_t)b * HWN + n0) * CCH;
  v8f acc0 = {}, acc1 = {};
#pragma unroll
  for (int k0 = 0; k0 < CCH; k0 += 32) {
    Frag fa  = load_afrag(A + k0, row, CCH, lane);
    Frag fb0 = load_bfrag(W + k0, row, CCH, lane);
    Frag fb1 = load_bfrag(W + (size_t)16 * CCH + k0, row, CCH, lane);
    acc0 = wmma_bf16(fa, fb0, acc0);
    acc1 = wmma_bf16(fa, fb1, acc1);
  }
  int o = lane & 15;
  int hi = (lane >> 4) & 1;
#pragma unroll
  for (int j = 0; j < 8; ++j) {
    int n = n0 + j + 8 * hi;
    __bf16* dp = outp + ((size_t)b * HWN + n) * DQK;
    dp[o]      = (__bf16)(acc0[j] + bias[o]);
    dp[o + 16] = (__bf16)(acc1[j] + bias[o + 16]);
  }
}

// ---------------- pass 1: V projection  Vp[b,c,m] = Wv[c,:] . yT[b,m,:] + bv[c]
__global__ __launch_bounds__(256) void v_proj_kernel(
    const __bf16* __restrict__ yT,   // [B, HW, 256]
    const __bf16* __restrict__ Wv,   // [256, 256]
    const float* __restrict__ bv,    // [256]
    __bf16* __restrict__ Vp)         // [B, 256, HW]
{
  int b = blockIdx.z;
  int ct = blockIdx.y;                 // c-tile of 16 rows
  int w = threadIdx.x >> 5, lane = threadIdx.x & 31;
  int m0 = blockIdx.x * 512 + w * 64;  // 4 m-tiles of 16 per wave
  int row = lane & 15;
  const __bf16* Ap = Wv + (size_t)ct * 16 * CCH;
  v8f acc[4] = {};
#pragma unroll
  for (int k0 = 0; k0 < CCH; k0 += 32) {
    Frag fa = load_afrag(Ap + k0, row, CCH, lane);
#pragma unroll
    for (int t = 0; t < 4; ++t) {
      Frag fb = load_bfrag(yT + ((size_t)b * HWN + m0 + t * 16) * CCH + k0, row, CCH, lane);
      acc[t] = wmma_bf16(fa, fb, acc[t]);
    }
  }
  int hi = (lane >> 4) & 1;
#pragma unroll
  for (int j = 0; j < 8; ++j) {
    int c = ct * 16 + j + 8 * hi;
    float bvc = bv[c];
#pragma unroll
    for (int t = 0; t < 4; ++t) {
      int m = m0 + t * 16 + (lane & 15);
      Vp[((size_t)b * CCH + c) * HWN + m] = (__bf16)(acc[t][j] + bvc);
    }
  }
}

// Build P as B-fragment [keys=32 x q=16] from two S^T D-tiles (exp'ed).
// lo lanes need keys 0..15  (0..7 local Plo, 8..15 from hi half's Plo)
// hi lanes need keys 16..31 (16..23 from lo half's Phi, 24..31 local Phi)
__device__ inline Frag build_pfrag(const float* Plo, const float* Phi, bool lo) {
  unsigned u0[4], u1[4];
#pragma unroll
  for (int j = 0; j < 4; ++j) {
    u0[j] = pack_bf16(Plo[2 * j], Plo[2 * j + 1]);
    u1[j] = pack_bf16(Phi[2 * j], Phi[2 * j + 1]);
  }
  Frag f;
#pragma unroll
  for (int j = 0; j < 4; ++j) {
    unsigned x0 = (unsigned)__shfl_xor((int)u0[j], 16, 32);
    unsigned x1 = (unsigned)__shfl_xor((int)u1[j], 16, 32);
    f.u[j]     = lo ? u0[j] : x1;
    f.u[j + 4] = lo ? x0 : u1[j];
  }
  return f;
}

// ---------------- pass 2: fused flash attention (double-buffered async LDS)
// block = 8 waves covering 128 queries; wave w owns 32 queries (2 tiles,
// group w>>1) x 128 channels (half w&1). Per 64-key step per wave:
// 8 S^T wmma (K-frags shared by both query tiles), 2 lane-local online
// softmaxes, 4 P-frags, 32 PV wmma with 2x reuse of every V fragment.
__global__ __launch_bounds__(256) void attn_kernel(
    const __bf16* __restrict__ Qp,   // [B, HW, 32]
    const __bf16* __restrict__ Kp,   // [B, HW, 32]  (K^T)
    const __bf16* __restrict__ Vp,   // [B, 256, HW]
    float* __restrict__ outp)        // [B, 256, HW]
{
  __shared__ __bf16 sK[2 * KC * SKLD];    // 2 x [64 keys x 32 d]
  __shared__ __bf16 sV[2 * CCH * SVLD];   // 2 x [256 ch x 64 keys]

  int b = blockIdx.y;
  int t = threadIdx.x;
  int w = t >> 5, lane = t & 31;
  int ch = (w & 1) * 128;                    // this wave's channel half
  int q0 = blockIdx.x * 256 + (w >> 1) * 32; // this wave's 32 queries
  int row = lane & 15;
  bool lo = (lane < 16);

  const __bf16* Vg = Vp + ((size_t)b * CCH + t) * HWN;  // this thread's V row
  const __bf16* Kg = Kp + (size_t)b * HWN * DQK;

  // Q as B-fragments [d=32 x q=16], constant across the key loop
  Frag fQ0 = load_bfrag(Qp + ((size_t)b * HWN + q0) * DQK, row, DQK, lane);
  Frag fQ1 = load_bfrag(Qp + ((size_t)b * HWN + q0 + 16) * DQK, row, DQK, lane);

  v8f z = {};
  v8f acc[2][8];
#pragma unroll
  for (int qt = 0; qt < 2; ++qt)
#pragma unroll
    for (int tl = 0; tl < 8; ++tl) acc[qt][tl] = z;
  float m_run[2] = {-3.0e38f, -3.0e38f};
  float l_run[2] = {0.f, 0.f};

  // stage chunk: per thread 8x16B of V (its channel row) + 1x16B of K
  auto stage = [&](int m0, int buf) {
    __bf16* vd = sV + buf * (CCH * SVLD) + t * SVLD;
    const __bf16* vs = Vg + m0;
#pragma unroll
    for (int j = 0; j < 8; ++j) cp16_g2l(vd + j * 8, vs + j * 8);
    // K: uint4 index = t -> row t>>2 (of 64), col (t&3)*8
    __bf16* kd = sK + buf * (KC * SKLD) + (t >> 2) * SKLD + (t & 3) * 8;
    cp16_g2l(kd, Kg + (size_t)(m0 + (t >> 2)) * DQK + (t & 3) * 8);
  };

  stage(0, 0);

  const int NSTEP = HWN / KC;  // 64
  for (int i = 0; i < NSTEP; ++i) {
    wait_async_done();
    __syncthreads();           // chunk i resident in buf (i&1); all reads of other buf done
    if (i + 1 < NSTEP) stage((i + 1) * KC, (i + 1) & 1);

    const __bf16* sKc = sK + (i & 1) * (KC * SKLD);
    const __bf16* sVc = sV + (i & 1) * (CCH * SVLD);

    // K fragments shared by both query tiles
    Frag fK[4];
#pragma unroll
    for (int t4 = 0; t4 < 4; ++t4)
      fK[t4] = load_afrag(sKc + t4 * 16 * SKLD, row, SKLD, lane);

    float scale[2];
    Frag fP[2][2];
#pragma unroll
    for (int qt = 0; qt < 2; ++qt) {
      const Frag& fq = qt ? fQ1 : fQ0;
      // S^T tiles: D[M=key, N=query]
      v8f S[4];
#pragma unroll
      for (int t4 = 0; t4 < 4; ++t4) S[t4] = wmma_bf16(fK[t4], fq, z);

      float mx = S[0][0];
#pragma unroll
      for (int t4 = 0; t4 < 4; ++t4)
#pragma unroll
        for (int j = 0; j < 8; ++j) mx = fmaxf(mx, S[t4][j]);
      mx = fmaxf(mx, __shfl_xor(mx, 16, 32));
      float m_new = fmaxf(m_run[qt], mx);
      scale[qt] = exp2f((m_run[qt] - m_new) * L2E);
      float P[4][8];
      float rs = 0.f;
#pragma unroll
      for (int t4 = 0; t4 < 4; ++t4)
#pragma unroll
        for (int j = 0; j < 8; ++j) {
          P[t4][j] = exp2f((S[t4][j] - m_new) * L2E);
          rs += P[t4][j];
        }
      rs += __shfl_xor(rs, 16, 32);
      l_run[qt] = l_run[qt] * scale[qt] + rs;
      m_run[qt] = m_new;

      fP[qt][0] = build_pfrag(P[0], P[1], lo);   // keys  0..31
      fP[qt][1] = build_pfrag(P[2], P[3], lo);   // keys 32..63
    }

    // O^T += V . P over 8 channel tiles; each V fragment feeds both q-tiles
#pragma unroll
    for (int tl = 0; tl < 8; ++tl) {
      const __bf16* vb = sVc + (size_t)(ch + tl * 16) * SVLD;
      Frag fVa = load_afrag(vb, row, SVLD, lane);
      Frag fVb = load_afrag(vb + 32, row, SVLD, lane);
#pragma unroll
      for (int qt = 0; qt < 2; ++qt) {
        v8f a = acc[qt][tl];
#pragma unroll
        for (int j = 0; j < 8; ++j) a[j] *= scale[qt];
        a = wmma_bf16(fVa, fP[qt][0], a);
        acc[qt][tl] = wmma_bf16(fVb, fP[qt][1], a);
      }
    }
  }

  int hi = (lane >> 4) & 1;
#pragma unroll
  for (int qt = 0; qt < 2; ++qt) {
    float inv = 1.0f / l_run[qt];   // per-lane == per-query
    int q = q0 + qt * 16 + (lane & 15);
#pragma unroll
    for (int tl = 0; tl < 8; ++tl)
#pragma unroll
      for (int j = 0; j < 8; ++j) {
        int c = ch + tl * 16 + j + 8 * hi;
        outp[((size_t)b * CCH + c) * HWN + q] = acc[qt][tl][j] * inv;
      }
  }
}

extern "C" void kernel_launch(void* const* d_in, const int* in_sizes, int n_in,
                              void* d_out, int out_size, void* d_ws, size_t ws_size,
                              hipStream_t stream) {
  const float* x  = (const float*)d_in[0];
  const float* y  = (const float*)d_in[1];
  const float* Wq = (const float*)d_in[2];
  const float* bq = (const float*)d_in[3];
  const float* Wk = (const float*)d_in[4];
  const float* bk = (const float*)d_in[5];
  const float* Wv = (const float*)d_in[6];
  const float* bv = (const float*)d_in[7];
  float* out = (float*)d_out;
  (void)in_sizes; (void)n_in; (void)out_size; (void)ws_size;

  const int B = 8;
  char* wsp = (char*)d_ws;
  size_t off = 0;
  auto alloc = [&](size_t bytes) {
    char* p = wsp + off;
    off += (bytes + 255) & ~(size_t)255;
    return p;
  };
  __bf16* xT  = (__bf16*)alloc((size_t)B * HWN * CCH * 2);
  __bf16* yT  = (__bf16*)alloc((size_t)B * HWN * CCH * 2);
  __bf16* Wqb = (__bf16*)alloc((size_t)DQK * CCH * 2);
  __bf16* Wkb = (__bf16*)alloc((size_t)DQK * CCH * 2);
  __bf16* Wvb = (__bf16*)alloc((size_t)CCH * CCH * 2);
  __bf16* Qp  = (__bf16*)alloc((size_t)B * HWN * DQK * 2);
  __bf16* Kp  = (__bf16*)alloc((size_t)B * HWN * DQK * 2);
  __bf16* Vp  = (__bf16*)alloc((size_t)B * CCH * HWN * 2);

  dim3 blk(256);
  transpose_cvt_kernel<<<dim3(HWN / 32, CCH / 32, B), blk, 0, stream>>>(x, xT, CCH, HWN);
  transpose_cvt_kernel<<<dim3(HWN / 32, CCH / 32, B), blk, 0, stream>>>(y, yT, CCH, HWN);
  cvt_bf16_kernel<<<(DQK * CCH + 255) / 256, blk, 0, stream>>>(Wq, Wqb, DQK * CCH);
  cvt_bf16_kernel<<<(DQK * CCH + 255) / 256, blk, 0, stream>>>(Wk, Wkb, DQK * CCH);
  cvt_bf16_kernel<<<(CCH * CCH + 255) / 256, blk, 0, stream>>>(Wv, Wvb, CCH * CCH);
  qk_proj_kernel<<<dim3(HWN / 128, B), blk, 0, stream>>>(xT, Wqb, bq, Qp);
  qk_proj_kernel<<<dim3(HWN / 128, B), blk, 0, stream>>>(yT, Wkb, bk, Kp);
  v_proj_kernel<<<dim3(HWN / 512, CCH / 16, B), blk, 0, stream>>>(yT, Wvb, bv, Vp);
  attn_kernel<<<dim3(HWN / 256, B), blk, 0, stream>>>(Qp, Kp, Vp, out);
}
